// GraphFeaturePyramid_75505525064738
// MI455X (gfx1250) — matrix operand order repaired and checked by reading, not compiled, not run on hardware
//
#include <hip/hip_runtime.h>
#include <hip/hip_bf16.h>

// ---------------------------------------------------------------------------
// GraphFeaturePyramid on MI455X (gfx1250)
//   - all matmuls on v_wmma_f32_16x16x32_f16 (f32 accumulate)
//   - B operand staged to LDS per 256-wide K chunk via GLOBAL_LOAD_ASYNC_TO_LDS
//   - each wave: 32 rows x 64 cols (8 accumulators, 2 A frags reused 4x)
// ---------------------------------------------------------------------------

typedef __attribute__((ext_vector_type(16))) _Float16 v16h;
typedef __attribute__((ext_vector_type(8)))  _Float16 v8h;
typedef __attribute__((ext_vector_type(8)))  float    v8f;
typedef __attribute__((ext_vector_type(4)))  int      v4i;

#if __has_builtin(__builtin_amdgcn_global_load_async_to_lds_b128)
#define HAVE_ASYNC_LDS 1
#endif

// 16B global -> LDS copy (async-tensor path; ASYNCcnt-tracked, bypasses VGPRs)
__device__ __forceinline__ void cp_b128_to_lds(const _Float16* g, _Float16* l) {
#if defined(HAVE_ASYNC_LDS)
  __builtin_amdgcn_global_load_async_to_lds_b128(
      (__attribute__((address_space(1))) v4i*)(g),
      (__attribute__((address_space(3))) v4i*)(l), 0, 0);
#else
  *(v8h*)l = *(const v8h*)g;
#endif
}

__device__ __forceinline__ void async_copies_wait() {
#if defined(HAVE_ASYNC_LDS)
#if __has_builtin(__builtin_amdgcn_s_wait_asynccnt)
  __builtin_amdgcn_s_wait_asynccnt(0);
#else
  asm volatile("s_wait_asynccnt 0x0" ::: "memory");
#endif
#endif
}

// A-fragment: per lane one contiguous run of 16 halves (k = 16h + s).
// p already includes the lane-half +16 offset.
__device__ __forceinline__ v16h a_frag_from(const _Float16* p) {
  v8h lo = *(const v8h*)p;
  v8h hi = *(const v8h*)(p + 8);
  v16h r;
#pragma unroll
  for (int i = 0; i < 8; ++i) { r[i] = lo[i]; r[i + 8] = hi[i]; }
  return r;
}

__device__ __forceinline__ v16h zero_v16h() {
  v16h z;
#pragma unroll
  for (int i = 0; i < 16; ++i) z[i] = (_Float16)0.0f;
  return z;
}

#define WMMA_F16(a, b, c) \
  __builtin_amdgcn_wmma_f32_16x16x32_f16(false, (a), false, (b), (short)0, (c), false, false)

// ---------------------------------------------------------------------------
// Generic WMMA GEMM:  C[M,Nc] = act(A[M,K] * Bt[Nc,K]^T + bias)
// Block = 256 threads = 8 waves; block tile 256 rows x 64 cols.
// Wave tile: 32 rows x 64 cols.  M is a multiple of 32 for every call site.
// B strip [64 x 256] staged in LDS per K chunk.
// ---------------------------------------------------------------------------
__global__ __launch_bounds__(256)
void gemm_wmma_f16(const _Float16* __restrict__ A, const _Float16* __restrict__ Bt,
                   const float* __restrict__ bias, float* __restrict__ C,
                   int M, int K, int Nc, int relu)
{
  __shared__ _Float16 bs[64 * 256];                 // 32 KB
  const int tid  = threadIdx.x;
  const int lane = tid & 31;
  const int wave = tid >> 5;
  const int colBase = blockIdx.y * 64;
  const int row0 = blockIdx.x * 256 + wave * 32;
  const bool active = (row0 < M);                   // wave-uniform
  const int nCol  = lane & 15;
  const int aHalf = (lane & 16) ? 16 : 0;           // A: lanes 16-31 hold K+16..31
  const int bOff  = (lane & 16) ? 16 : 0;           // B: lanes 16-31 hold K+16..31

  const size_t aRow0 = (size_t)(row0 + nCol) * K + aHalf;
  const size_t aRow1 = aRow0 + (size_t)16 * K;

  v8f acc[2][4] = {};

  for (int kc = 0; kc < K; kc += 256) {
    // ---- stage Bt[colBase..+63][kc..+255] into LDS: 8 x 16B per thread ----
#pragma unroll
    for (int i = 0; i < 8; ++i) {
      const int u   = tid + i * 256;                // 16-byte unit id, 0..2047
      const int col = u >> 5;                       // /32 units per column
      const int kk  = (u & 31) << 3;                // halfs within column
      cp_b128_to_lds(Bt + (size_t)(colBase + col) * K + kc + kk, &bs[col * 256 + kk]);
    }
    async_copies_wait();
    __syncthreads();

    if (active) {
      for (int kt = 0; kt < 256; kt += 32) {
        __builtin_prefetch(A + aRow0 + kc + kt + 256, 0, 1);    // global_prefetch_b8
        const v16h a0 = a_frag_from(A + aRow0 + kc + kt);
        const v16h a1 = a_frag_from(A + aRow1 + kc + kt);
#pragma unroll
        for (int t = 0; t < 4; ++t) {
          const v16h b = *(const v16h*)(&bs[(t * 16 + nCol) * 256 + kt + bOff]);
          acc[0][t] = WMMA_F16(a0, b, acc[0][t]);
          acc[1][t] = WMMA_F16(a1, b, acc[1][t]);
        }
      }
    }
    __syncthreads();
  }
  if (!active) return;

#pragma unroll
  for (int rt = 0; rt < 2; ++rt) {
    const int mBase = row0 + rt * 16 + ((lane & 16) ? 8 : 0);   // C/D: lanes16-31 -> M+8
#pragma unroll
    for (int t = 0; t < 4; ++t) {
      const int col = colBase + t * 16 + nCol;
      const float bb = bias ? bias[col] : 0.0f;
#pragma unroll
      for (int r = 0; r < 8; ++r) {
        float v = acc[rt][t][r] + bb;
        if (relu) v = fmaxf(v, 0.0f);
        C[(size_t)(mBase + r) * Nc + col] = v;
      }
    }
  }
}

// ---------------------------------------------------------------------------
// Implicit-GEMM 3x3 conv:  out[NCHW] = X[NHWC f16] (*) W + bias
// K = 9*256; each 256-wide K chunk is exactly one (dy,dx) filter tap, so the
// halo test is hoisted out of the inner loop.  B tap-slab staged in LDS.
// ---------------------------------------------------------------------------
__global__ __launch_bounds__(256)
void conv3x3_wmma_f16(const _Float16* __restrict__ X, const _Float16* __restrict__ Bt,
                      const float* __restrict__ bias, float* __restrict__ out,
                      int Bn, int H, int W)
{
  __shared__ _Float16 bs[64 * 256];
  const int tid  = threadIdx.x;
  const int lane = tid & 31;
  const int wave = tid >> 5;
  const int HW = H * W;
  const int M = Bn * HW;
  const int colBase = blockIdx.y * 64;
  const int row0 = blockIdx.x * 256 + wave * 32;
  const bool active = (row0 < M);
  const int nCol = lane & 15;
  const int aSub = (lane & 16) ? 16 : 0;
  const int bOff = (lane & 16) ? 16 : 0;
  const int K = 2304;

  // this lane's two A-tile pixels (rows m and m+16 of the wave tile)
  int pb[2], py[2], px[2];
#pragma unroll
  for (int rt = 0; rt < 2; ++rt) {
    const int m = (active ? row0 : 0) + rt * 16 + nCol;
    const int b = m / HW;
    const int rem = m - b * HW;
    const int y = rem / W;
    pb[rt] = b; py[rt] = y; px[rt] = rem - y * W;
  }

  v8f acc[2][4] = {};

  for (int tap = 0; tap < 9; ++tap) {
    const int kc = tap << 8;
    // ---- stage Bt[colBase..+63][tap*256 .. +255] into LDS ----
#pragma unroll
    for (int i = 0; i < 8; ++i) {
      const int u   = tid + i * 256;
      const int col = u >> 5;
      const int kk  = (u & 31) << 3;
      cp_b128_to_lds(Bt + (size_t)(colBase + col) * K + kc + kk, &bs[col * 256 + kk]);
    }
    async_copies_wait();
    __syncthreads();

    if (active) {
      const int dy = tap / 3 - 1;
      const int dx = tap - (tap / 3) * 3 - 1;
      const _Float16* ap[2];
      bool ok[2];
#pragma unroll
      for (int rt = 0; rt < 2; ++rt) {
        const int yy = py[rt] + dy, xx = px[rt] + dx;
        ok[rt] = ((unsigned)yy < (unsigned)H) && ((unsigned)xx < (unsigned)W);
        ap[rt] = X + ((size_t)(pb[rt] * HW + yy * W + xx) << 8) + aSub;
      }
      for (int kt = 0; kt < 256; kt += 32) {
        const v16h a0 = ok[0] ? a_frag_from(ap[0] + kt) : zero_v16h();
        const v16h a1 = ok[1] ? a_frag_from(ap[1] + kt) : zero_v16h();
#pragma unroll
        for (int t = 0; t < 4; ++t) {
          const v16h b = *(const v16h*)(&bs[(t * 16 + nCol) * 256 + kt + bOff]);
          acc[0][t] = WMMA_F16(a0, b, acc[0][t]);
          acc[1][t] = WMMA_F16(a1, b, acc[1][t]);
        }
      }
    }
    __syncthreads();
  }
  if (!active) return;

#pragma unroll
  for (int rt = 0; rt < 2; ++rt) {
    const int mBase = row0 + rt * 16 + ((lane & 16) ? 8 : 0);
#pragma unroll
    for (int t = 0; t < 4; ++t) {
      const int col = colBase + t * 16 + nCol;
      const float bb = bias[col];
#pragma unroll
      for (int r = 0; r < 8; ++r) {
        const int mm = mBase + r;
        const int bo = mm / HW;
        const int hw = mm - bo * HW;
        out[((size_t)bo * 256 + col) * HW + hw] = acc[rt][t][r] + bb;   // NCHW
      }
    }
  }
}

// ---------------------------------------------------------------------------
// LDS-tiled transpose + f32->f16 convert: in[R,Cc] (per batch) -> out[Cc,R]
// ---------------------------------------------------------------------------
__global__ void transpose_cvt_f16(const float* __restrict__ in, _Float16* __restrict__ out,
                                  int R, int Cc, long inBatchStride, long outBatchStride)
{
  __shared__ float tile[32][33];
  const int bz = blockIdx.z;
  const float* I = in + (size_t)bz * inBatchStride;
  _Float16* O = out + (size_t)bz * outBatchStride;
  const int c0 = blockIdx.x * 32;
  const int r0 = blockIdx.y * 32;
  const int tx = threadIdx.x, ty = threadIdx.y;
  for (int i = ty; i < 32; i += 8) {
    const int r = r0 + i, c = c0 + tx;
    tile[i][tx] = (r < R && c < Cc) ? I[(size_t)r * Cc + c] : 0.0f;
  }
  __syncthreads();
  for (int i = ty; i < 32; i += 8) {
    const int oc = c0 + i;      // out row  = in col
    const int orw = r0 + tx;    // out col  = in row
    if (oc < Cc && orw < R) O[(size_t)oc * R + orw] = (_Float16)tile[tx][i];
  }
}

__global__ void cvt_f16_k(const float* __restrict__ in, _Float16* __restrict__ out, long n) {
  const long t = (long)blockIdx.x * blockDim.x + threadIdx.x;
  if (t < n) out[t] = (_Float16)in[t];
}

// w[o,c,dy,dx] -> Bt[o][(dy*3+dx)*256 + c]  (f16)
__global__ void reorder_w3x3_k(const float* __restrict__ w, _Float16* __restrict__ out) {
  const long t = (long)blockIdx.x * blockDim.x + threadIdx.x;
  if (t >= 256L * 2304L) return;
  const int o = (int)(t / 2304);
  const int k = (int)(t - (long)o * 2304);
  const int tap = k >> 8;
  const int c = k & 255;
  const int dy = tap / 3, dx = tap - dy * 3;
  out[t] = (_Float16)w[((size_t)(o * 256 + c) * 3 + dy) * 3 + dx];
}

__global__ void degree_k(const int* __restrict__ dst, int E, float* __restrict__ deg) {
  const int t = blockIdx.x * blockDim.x + threadIdx.x;
  if (t < E) atomicAdd(&deg[dst[t]], 1.0f);
}

// agg[dst] += h[src]; 64 threads/edge, float4 per thread, L2-resident atomics
__global__ void scatter_k(const float* __restrict__ h, const int* __restrict__ src,
                          const int* __restrict__ dst, int E, float* __restrict__ agg) {
  const long t = (long)blockIdx.x * blockDim.x + threadIdx.x;
  const long e = t >> 6;
  if (e >= E) return;
  const int c0 = (int)(t & 63) << 2;
  const int s = src[e], d = dst[e];
  const float4 v = *(const float4*)(h + ((size_t)s << 8) + c0);
  float* ap = agg + ((size_t)d << 8) + c0;
  atomicAdd(ap + 0, v.x); atomicAdd(ap + 1, v.y);
  atomicAdd(ap + 2, v.z); atomicAdd(ap + 3, v.w);
}

// x = f16(h + agg / max(deg,1))
__global__ void build_x_k(const float* __restrict__ h, const float* __restrict__ agg,
                          const float* __restrict__ deg, _Float16* __restrict__ x, long Nn) {
  const long t = (long)blockIdx.x * blockDim.x + threadIdx.x;
  if (t >= Nn * 64) return;
  const long n = t >> 6;
  const int c0 = (int)(t & 63) << 2;
  const float dg = fmaxf(deg[n], 1.0f);
  const float4 hv = *(const float4*)(h + (n << 8) + c0);
  const float4 av = *(const float4*)(agg + (n << 8) + c0);
  _Float16* xp = x + (n << 8) + c0;
  xp[0] = (_Float16)(hv.x + av.x / dg);
  xp[1] = (_Float16)(hv.y + av.y / dg);
  xp[2] = (_Float16)(hv.z + av.z / dg);
  xp[3] = (_Float16)(hv.w + av.w / dg);
}

// p = p_lateral + h_final [+ up2(parent)] ; emits f32 (next level) + f16 (conv in)
__global__ void combine_k(const float* __restrict__ p0, const float* __restrict__ hg,
                          const float* __restrict__ parent, float* __restrict__ pout,
                          _Float16* __restrict__ pout16, int Bn, int H, int W) {
  const long t = (long)blockIdx.x * blockDim.x + threadIdx.x;
  const long total = (long)Bn * H * W * 256;
  if (t >= total) return;
  const long n = t >> 8;
  const int c = (int)(t & 255);
  float v = p0[t] + hg[t];
  if (parent) {
    const int Wp = W >> 1;
    const int b = (int)(n / (H * W));
    const int rem = (int)(n - (long)b * H * W);
    const int y = rem / W, x = rem - y * W;
    const long pn = (long)b * (H >> 1) * Wp + (long)(y >> 1) * Wp + (x >> 1);
    v += parent[(pn << 8) + c];
  }
  pout[t] = v;
  pout16[t] = (_Float16)v;
}

// ---------------------------------------------------------------------------
static inline unsigned cdiv(long a, long b) { return (unsigned)((a + b - 1) / b); }

extern "C" void kernel_launch(void* const* d_in, const int* in_sizes, int n_in,
                              void* d_out, int out_size, void* d_ws, size_t ws_size,
                              hipStream_t stream) {
  (void)n_in; (void)out_size; (void)ws_size;
  const long N3 = 25600, N4 = 6400, N5 = 1600, NN = 33600, BS = 4;
  const long OFF4 = N3, OFF5 = N3 + N4;

  const float* c3 = (const float*)d_in[0];
  const float* c4 = (const float*)d_in[1];
  const float* c5 = (const float*)d_in[2];
  const float* w3_1 = (const float*)d_in[3];  const float* b3_1 = (const float*)d_in[4];
  const float* w4_1 = (const float*)d_in[5];  const float* b4_1 = (const float*)d_in[6];
  const float* w5_1 = (const float*)d_in[7];  const float* b5_1 = (const float*)d_in[8];
  const float* w3_3 = (const float*)d_in[9];  const float* b3_3 = (const float*)d_in[10];
  const float* w4_3 = (const float*)d_in[11]; const float* b4_3 = (const float*)d_in[12];
  const float* w5_3 = (const float*)d_in[13]; const float* b5_3 = (const float*)d_in[14];
  const float* Wc = (const float*)d_in[15];   const float* bc = (const float*)d_in[16];
  const float* Wh = (const float*)d_in[17];   const float* bh = (const float*)d_in[18];
  const int* src_c = (const int*)d_in[19];    const int* dst_c = (const int*)d_in[20];
  const int* src_h = (const int*)d_in[21];    const int* dst_h = (const int*)d_in[22];
  const int EC = in_sizes[19], EH = in_sizes[21];
  float* out = (float*)d_out;

  // ---- carve workspace (256B-aligned slabs) ------------------------------
  char* wp = (char*)d_ws;
  auto carve = [&](size_t bytes) -> void* {
    void* p = (void*)wp; wp += (bytes + 255) & ~(size_t)255; return p;
  };
  float*    h0   = (float*)   carve(NN * 256 * 4);   // lateral outputs / initial nodes
  float*    hA   = (float*)   carve(NN * 256 * 4);   // evolving node features
  float*    agg  = (float*)   carve(NN * 256 * 4);   // scatter target, later p_new f32
  float*    degc = (float*)   carve(NN * 4);
  float*    degh = (float*)   carve(NN * 4);
  _Float16* x16  = (_Float16*)carve(NN * 256 * 2);   // GNN GEMM input, later p f16
  _Float16* A3   = (_Float16*)carve(N3 * 256 * 2);
  _Float16* A4   = (_Float16*)carve(N4 * 512 * 2);
  _Float16* A5   = (_Float16*)carve(N5 * 512 * 2);
  _Float16* w3l  = (_Float16*)carve(256 * 256 * 2);
  _Float16* w4l  = (_Float16*)carve(256 * 512 * 2);
  _Float16* w5l  = (_Float16*)carve(256 * 512 * 2);
  _Float16* wct  = (_Float16*)carve(6 * 256 * 256 * 2);
  _Float16* wht  = (_Float16*)carve(3 * 256 * 256 * 2);
  _Float16* B33  = (_Float16*)carve(2304 * 256 * 2);
  _Float16* B43  = (_Float16*)carve(2304 * 256 * 2);
  _Float16* B53  = (_Float16*)carve(2304 * 256 * 2);

  const dim3 tb(32, 8, 1);
  // NCHW -> [pix, chan] f16 (per-batch transposes)
  transpose_cvt_f16<<<dim3(cdiv(6400, 32), cdiv(256, 32), 4), tb, 0, stream>>>(c3, A3, 256, 6400, 256L * 6400, 6400L * 256);
  transpose_cvt_f16<<<dim3(cdiv(1600, 32), cdiv(512, 32), 4), tb, 0, stream>>>(c4, A4, 512, 1600, 512L * 1600, 1600L * 512);
  transpose_cvt_f16<<<dim3(cdiv(400, 32),  cdiv(512, 32), 4), tb, 0, stream>>>(c5, A5, 512, 400,  512L * 400,  400L * 512);
  // GNN weights: [k,n] -> [n,k] f16
  transpose_cvt_f16<<<dim3(8, 8, 6), tb, 0, stream>>>(Wc, wct, 256, 256, 65536L, 65536L);
  transpose_cvt_f16<<<dim3(8, 8, 3), tb, 0, stream>>>(Wh, wht, 256, 256, 65536L, 65536L);
  // 1x1 weights already [o,k]
  cvt_f16_k<<<cdiv(65536, 256), 256, 0, stream>>>(w3_1, w3l, 65536);
  cvt_f16_k<<<cdiv(131072, 256), 256, 0, stream>>>(w4_1, w4l, 131072);
  cvt_f16_k<<<cdiv(131072, 256), 256, 0, stream>>>(w5_1, w5l, 131072);
  // 3x3 weights -> [o][tap*256+c]
  reorder_w3x3_k<<<cdiv(256L * 2304, 256), 256, 0, stream>>>(w3_3, B33);
  reorder_w3x3_k<<<cdiv(256L * 2304, 256), 256, 0, stream>>>(w4_3, B43);
  reorder_w3x3_k<<<cdiv(256L * 2304, 256), 256, 0, stream>>>(w5_3, B53);

  // lateral 1x1 convs -> h0 (concatenated node features, NHWC order)
  gemm_wmma_f16<<<dim3(cdiv(N3, 256), 4), 256, 0, stream>>>(A3, w3l, b3_1, h0,              (int)N3, 256, 256, 0);
  gemm_wmma_f16<<<dim3(cdiv(N4, 256), 4), 256, 0, stream>>>(A4, w4l, b4_1, h0 + OFF4 * 256, (int)N4, 512, 256, 0);
  gemm_wmma_f16<<<dim3(cdiv(N5, 256), 4), 256, 0, stream>>>(A5, w5l, b5_1, h0 + OFF5 * 256, (int)N5, 512, 256, 0);

  // degrees
  (void)hipMemsetAsync(degc, 0, NN * 4, stream);
  (void)hipMemsetAsync(degh, 0, NN * 4, stream);
  degree_k<<<cdiv(EC, 256), 256, 0, stream>>>(dst_c, EC, degc);
  degree_k<<<cdiv(EH, 256), 256, 0, stream>>>(dst_h, EH, degh);

  // 9 GNN layers: c0,c1,c2, h0,h1,h2, c3,c4,c5
  const float* cur = h0;
  for (int L = 0; L < 9; ++L) {
    const bool hier = (L >= 3 && L < 6);
    const int wi = hier ? (L - 3) : (L < 3 ? L : L - 3);
    const _Float16* Wt  = hier ? (wht + (size_t)wi * 65536) : (wct + (size_t)wi * 65536);
    const float*    bW  = hier ? (bh + wi * 256) : (bc + wi * 256);
    const int*      src = hier ? src_h : src_c;
    const int*      dst = hier ? dst_h : dst_c;
    const float*    deg = hier ? degh : degc;
    const int       E   = hier ? EH : EC;

    (void)hipMemsetAsync(agg, 0, NN * 256 * 4, stream);
    scatter_k<<<cdiv((long)E * 64, 256), 256, 0, stream>>>(cur, src, dst, E, agg);
    build_x_k<<<cdiv(NN * 64, 256), 256, 0, stream>>>(cur, agg, deg, x16, NN);
    gemm_wmma_f16<<<dim3(cdiv(NN, 256), 4), 256, 0, stream>>>(x16, Wt, bW, hA, (int)NN, 256, 256, 1);
    cur = hA;
  }

  // top-down combine (p_new f32 into agg, f16 into x16); order: P5 -> P4 -> P3
  combine_k<<<cdiv(N5 * 256, 256), 256, 0, stream>>>(h0 + OFF5 * 256, hA + OFF5 * 256, nullptr,
                                                     agg + OFF5 * 256, x16 + OFF5 * 256, (int)BS, 20, 20);
  combine_k<<<cdiv(N4 * 256, 256), 256, 0, stream>>>(h0 + OFF4 * 256, hA + OFF4 * 256, agg + OFF5 * 256,
                                                     agg + OFF4 * 256, x16 + OFF4 * 256, (int)BS, 40, 40);
  combine_k<<<cdiv(N3 * 256, 256), 256, 0, stream>>>(h0, hA, agg + OFF4 * 256,
                                                     agg, x16, (int)BS, 80, 80);

  // 3x3 output convs (implicit GEMM, NCHW f32 out)
  float* out3 = out;
  float* out4 = out + (size_t)BS * 256 * 80 * 80;
  float* out5 = out4 + (size_t)BS * 256 * 40 * 40;
  conv3x3_wmma_f16<<<dim3(cdiv(N3, 256), 4), 256, 0, stream>>>(x16,              B33, b3_3, out3, (int)BS, 80, 80);
  conv3x3_wmma_f16<<<dim3(cdiv(N4, 256), 4), 256, 0, stream>>>(x16 + OFF4 * 256, B43, b4_3, out4, (int)BS, 40, 40);
  conv3x3_wmma_f16<<<dim3(cdiv(N5, 256), 4), 256, 0, stream>>>(x16 + OFF5 * 256, B53, b5_3, out5, (int)BS, 20, 20);
}
